// PointTransformerV1_77017353552245
// MI455X (gfx1250) — compile-verified, matrix operand order checked by reading, not run on hardware
//
#include <hip/hip_runtime.h>
#include <hip/hip_bf16.h>

// ---------------------------------------------------------------------------
// PointTransformerV1 fused attention for MI455X (gfx1250, wave32, WMMA).
//   N=16384 points, C=256 channels, G=16 groups, S=16 neighbors.
// Kernel 0: pack fp32 weights into bf16 WMMA B-fragment order (one-time,
//           ~0.5 MB, L2-resident): each lane's fragment is one 32B chunk.
// Kernel 1: q/k/v projections, 32 rows/block; two A tiles share every
//           B fragment (2 WMMAs per B load). k is stored bf16 (it is only
//           consumed through the bf16 rel operand downstream).
// Kernel 2: fused attention, 2 points/block; gathers neighbors to LDS,
//           runs the 16x256x256 peb GEMM and 16x16x256 weight GEMM on
//           v_wmma_f32_16x16x32_bf16 with double-buffered B fragments,
//           then softmax + grouped weighted reduction. Nothing of size
//           [N,S,C] ever touches HBM.
// ---------------------------------------------------------------------------

#define NPTS 16384
#define CCH  256
#define GRP  16
#define SNB  16

typedef __attribute__((ext_vector_type(16))) __bf16 v16bf;
typedef __attribute__((ext_vector_type(8)))  __bf16 v8bf;
typedef __attribute__((ext_vector_type(8)))  float  v8f;

static __device__ __forceinline__ __bf16 tobf(float x) { return (__bf16)x; }

// Load a 16x32 A-fragment row slice from LDS: two 16B-aligned 8-element
// runs (ds_load_b128) combined with a register-only shuffle.
static __device__ __forceinline__ v16bf load_a_frag(const __bf16* rowp, int kA) {
    v8bf lo = *(const v8bf*)(rowp + kA);
    v8bf hh = *(const v8bf*)(rowp + kA + 16);
    return __builtin_shufflevector(lo, hh, 0, 1, 2, 3, 4, 5, 6, 7,
                                            8, 9, 10, 11, 12, 13, 14, 15);
}

// ---------------------------------------------------------------------------
// Kernel 0: pack W[k][n] (K=256 rows, ncols columns, fp32 row-major) into
// bf16 B-fragment order: P[(((ct*8)+kk)*32 + lane)*16 + e] = W[k][n]
//   k = kk*32 + (e>=8 ? 16 : 0) + (lane>>4)*8 + (e&7),  n = ct*16 + (lane&15)
// ---------------------------------------------------------------------------
__global__ __launch_bounds__(256) void pack_weight_kernel(
    const float* __restrict__ W, __bf16* __restrict__ P, int ncols)
{
    const int tid = blockIdx.x * 256 + threadIdx.x;
    if (tid >= 256 * ncols) return;
    const int e    = tid & 15;
    const int lane = (tid >> 4) & 31;
    const int kk   = (tid >> 9) & 7;
    const int ct   = tid >> 12;
    const int k = kk * 32 + ((e & 8) ? 16 : 0) + (lane >> 4) * 8 + (e & 7);
    const int n = ct * 16 + (lane & 15);
    P[tid] = tobf(W[(size_t)k * ncols + n]);
}

// ---------------------------------------------------------------------------
// Kernel 1: q/k/v projections, 32-row blocks (two 16-row A tiles).
//   q = relu((feat@Wq + bq)*qs + qh) ; k likewise (stored bf16) ; v = feat@Wv + bv
// ---------------------------------------------------------------------------
__global__ __launch_bounds__(256) void qkv_gemm_kernel(
    const float* __restrict__ feat,
    const __bf16* __restrict__ pWq, const __bf16* __restrict__ pWk,
    const __bf16* __restrict__ pWv,
    const float* __restrict__ bq, const float* __restrict__ qsc, const float* __restrict__ qsh,
    const float* __restrict__ bk, const float* __restrict__ ksc, const float* __restrict__ ksh,
    const float* __restrict__ bv,
    float* __restrict__ qo, __bf16* __restrict__ ko, float* __restrict__ vo)
{
    __shared__ __bf16 Atile[2][16][CCH];        // 16 KB bf16 activation tiles

    const int tid  = threadIdx.x;
    const int row0 = blockIdx.x * 32;

    // Cooperative load of the 32x256 fp32 activation tile -> bf16 LDS.
#pragma unroll
    for (int p = 0; p < 2; ++p) {
        const int r  = tid >> 4;
        const int c0 = (tid & 15) * 16;
        const float4* src = (const float4*)(feat + (size_t)(row0 + p * 16 + r) * CCH + c0);
#pragma unroll
        for (int i = 0; i < 4; ++i) {
            float4 f = src[i];
            Atile[p][r][c0 + i * 4 + 0] = tobf(f.x);
            Atile[p][r][c0 + i * 4 + 1] = tobf(f.y);
            Atile[p][r][c0 + i * 4 + 2] = tobf(f.z);
            Atile[p][r][c0 + i * 4 + 3] = tobf(f.w);
        }
    }
    __syncthreads();

    const int wave = tid >> 5;
    const int lane = tid & 31;
    const int mrow = lane & 15;      // A row / B column within tile
    const int hi   = lane >> 4;      // half-wave -> K striping

    for (int t = 0; t < 6; ++t) {
        const int ct       = wave * 6 + t;      // 0..47 column tile
        const int mat      = ct >> 4;           // 0=q 1=k 2=v
        const int col_base = (ct & 15) * 16;
        const __bf16* Pm = (mat == 0) ? pWq : (mat == 1) ? pWk : pWv;
        const v16bf* Bt  = (const v16bf*)Pm + (size_t)(ct & 15) * 8 * 32;

        v8f accP[2] = {};
        v16bf bcur = Bt[lane];                  // double-buffered B stream
#pragma unroll
        for (int kk = 0; kk < 8; ++kk) {        // K = 256 in steps of 32
            v16bf bnext = bcur;
            if (kk < 7) bnext = Bt[(kk + 1) * 32 + lane];
            if (kk < 6) __builtin_prefetch((const void*)(Bt + (kk + 2) * 32 + lane), 0, 3);
            const int kA = kk * 32 + hi * 8;
            v16bf a0 = load_a_frag(&Atile[0][mrow][0], kA);
            v16bf a1 = load_a_frag(&Atile[1][mrow][0], kA);
            accP[0] = __builtin_amdgcn_wmma_f32_16x16x32_bf16(
                false, a0, false, bcur, (short)0, accP[0], false, false);
            accP[1] = __builtin_amdgcn_wmma_f32_16x16x32_bf16(
                false, a1, false, bcur, (short)0, accP[1], false, false);
            bcur = bnext;
        }

        // Epilogue: per-column bias (+ folded BN scale/shift + relu for q,k).
        const int ncol = col_base + mrow;
        float bias, sc = 1.f, sh = 0.f;
        if (mat == 0)      { bias = bq[ncol]; sc = qsc[ncol]; sh = qsh[ncol]; }
        else if (mat == 1) { bias = bk[ncol]; sc = ksc[ncol]; sh = ksh[ncol]; }
        else               { bias = bv[ncol]; }
#pragma unroll
        for (int p = 0; p < 2; ++p) {
#pragma unroll
            for (int r = 0; r < 8; ++r) {
                const int m = p * 16 + r + hi * 8;
                float x = accP[p][r] + bias;
                if (mat != 2) { x = x * sc + sh; x = x > 0.f ? x : 0.f; }
                if (mat == 0)      qo[(size_t)(row0 + m) * CCH + ncol] = x;
                else if (mat == 1) ko[(size_t)(row0 + m) * CCH + ncol] = tobf(x);
                else               vo[(size_t)(row0 + m) * CCH + ncol] = x;
            }
        }
    }
}

// ---------------------------------------------------------------------------
// Kernel 2: fused attention, 2 points per block. grid.x = N/2, block = 256.
// ---------------------------------------------------------------------------
__global__ __launch_bounds__(256) void attn_fused_kernel(
    const float* __restrict__ qbuf, const __bf16* __restrict__ kbuf,
    const float* __restrict__ vbuf, const float* __restrict__ coord,
    const float* __restrict__ Wp1, const float* __restrict__ bp1,
    const float* __restrict__ psc, const float* __restrict__ psh,
    const __bf16* __restrict__ pWp2, const float* __restrict__ bp2,
    const __bf16* __restrict__ pWw1, const float* __restrict__ bw1,
    const float* __restrict__ wsc, const float* __restrict__ wsh,
    const float* __restrict__ Ww2, const float* __restrict__ bw2,
    const int* __restrict__ refidx,
    float* __restrict__ out)
{
    __shared__ float  qrow[2][CCH];         // 2 KB
    __shared__ __bf16 kgb[2][SNB][CCH];     // 16 KB gathered k (bf16)
    __shared__ float  vg[2][SNB][CCH];      // 32 KB gathered v (then +peb)
    __shared__ __bf16 h1[2][SNB][CCH];      // 16 KB relu(pos@Wp1...) bf16
    __shared__ __bf16 rel[2][SNB][CCH];     // 16 KB (kg - q + peb) bf16
    __shared__ float  posS[2][SNB][4];
    __shared__ int    idxS[2][SNB];
    __shared__ float  maskS[2][SNB];
    __shared__ float  wg1[2][SNB][GRP];
    __shared__ float  wg2[2][SNB][GRP];

    const int n0  = blockIdx.x * 2;
    const int tid = threadIdx.x;

    // ---- indices, relative positions, q rows ----
    if (tid < 2 * SNB) {
        const int p = tid >> 4, s = tid & 15;
        const int n = n0 + p;
        int id = refidx[(size_t)n * SNB + s];
        idxS[p][s]  = id < 0 ? 0 : id;
        maskS[p][s] = (id + 1) > 0 ? 1.f : 0.f;      // sign(idx+1)
        const int ic = id < 0 ? 0 : id;
        posS[p][s][0] = coord[(size_t)ic * 3 + 0] - coord[(size_t)n * 3 + 0];
        posS[p][s][1] = coord[(size_t)ic * 3 + 1] - coord[(size_t)n * 3 + 1];
        posS[p][s][2] = coord[(size_t)ic * 3 + 2] - coord[(size_t)n * 3 + 2];
    }
    qrow[0][tid] = qbuf[(size_t)n0 * CCH + tid];
    qrow[1][tid] = qbuf[(size_t)(n0 + 1) * CCH + tid];
    __syncthreads();

    // ---- gather k (bf16, raw 16B copies) and v (fp32) neighbor rows ----
#pragma unroll
    for (int p = 0; p < 2; ++p) {
        const int r  = tid >> 4;
        const int c0 = (tid & 15) * 16;
        const int src = idxS[p][r];
        const v8bf*   ks = (const v8bf*)(kbuf + (size_t)src * CCH + c0);
        const float4* vs = (const float4*)(vbuf + (size_t)src * CCH + c0);
        v8bf*   kd = (v8bf*)(&kgb[p][r][c0]);
        float4* vd = (float4*)(&vg[p][r][c0]);
        kd[0] = ks[0];
        kd[1] = ks[1];
#pragma unroll
        for (int i = 0; i < 4; ++i) vd[i] = vs[i];
    }

    // ---- h1 = relu((pos @ Wp1 + bp1)*psc + psh) : K=3, VALU ----
    {
        const float w0 = Wp1[0 * CCH + tid];
        const float w1 = Wp1[1 * CCH + tid];
        const float w2 = Wp1[2 * CCH + tid];
        const float b  = bp1[tid], sc = psc[tid], sh = psh[tid];
#pragma unroll
        for (int p = 0; p < 2; ++p)
#pragma unroll
            for (int s = 0; s < SNB; ++s) {
                float x = posS[p][s][0] * w0 + posS[p][s][1] * w1 +
                          posS[p][s][2] * w2 + b;
                x = x * sc + sh;
                h1[p][s][tid] = tobf(x > 0.f ? x : 0.f);
            }
    }
    __syncthreads();

    const int wave = tid >> 5;
    const int lane = tid & 31;
    const int mrow = lane & 15;
    const int hi   = lane >> 4;

    // ---- peb = h1 @ Wp2 + bp2 (16x256x256 per point, WMMA bf16) ----
    // Each B fragment (double-buffered) feeds both points. Fused epilogue:
    //   rel = kg - q + peb (->bf16 LDS), vg += peb.
    for (int t = 0; t < 2; ++t) {
        const int ctile    = wave * 2 + t;
        const int col_base = ctile * 16;
        const v16bf* Bt = (const v16bf*)pWp2 + (size_t)ctile * 8 * 32;

        v8f accP[2] = {};
        v16bf bcur = Bt[lane];
#pragma unroll
        for (int kk = 0; kk < 8; ++kk) {
            v16bf bnext = bcur;
            if (kk < 7) bnext = Bt[(kk + 1) * 32 + lane];
            if (kk < 6) __builtin_prefetch((const void*)(Bt + (kk + 2) * 32 + lane), 0, 3);
            const int kA = kk * 32 + hi * 8;
            v16bf a0 = load_a_frag(&h1[0][mrow][0], kA);
            v16bf a1 = load_a_frag(&h1[1][mrow][0], kA);
            accP[0] = __builtin_amdgcn_wmma_f32_16x16x32_bf16(
                false, a0, false, bcur, (short)0, accP[0], false, false);
            accP[1] = __builtin_amdgcn_wmma_f32_16x16x32_bf16(
                false, a1, false, bcur, (short)0, accP[1], false, false);
            bcur = bnext;
        }
        const int ncol = col_base + mrow;
        const float b2 = bp2[ncol];
#pragma unroll
        for (int p = 0; p < 2; ++p) {
            const float qv = qrow[p][ncol];
#pragma unroll
            for (int r = 0; r < 8; ++r) {
                const int s = r + hi * 8;            // neighbor (M) index
                const float peb = accP[p][r] + b2;
                rel[p][s][ncol] = tobf((float)kgb[p][s][ncol] - qv + peb);
                vg[p][s][ncol]  = vg[p][s][ncol] + peb;
            }
        }
    }
    __syncthreads();

    // ---- wg1 = relu((rel @ Ww1 + bw1)*wsc + wsh) : 16x16x256 WMMA ----
    // wave 0 -> point 0, wave 1 -> point 1.
    if (wave < 2) {
        const int p = wave;
        const v16bf* Bt = (const v16bf*)pWw1;
        v8f acc = {};
        v16bf bcur = Bt[lane];
#pragma unroll
        for (int kk = 0; kk < 8; ++kk) {
            v16bf bnext = bcur;
            if (kk < 7) bnext = Bt[(kk + 1) * 32 + lane];
            const int kA = kk * 32 + hi * 8;
            v16bf a = load_a_frag(&rel[p][mrow][0], kA);
            acc = __builtin_amdgcn_wmma_f32_16x16x32_bf16(
                false, a, false, bcur, (short)0, acc, false, false);
            bcur = bnext;
        }
        const int g = mrow;
        const float b1 = bw1[g], sc = wsc[g], sh = wsh[g];
#pragma unroll
        for (int r = 0; r < 8; ++r) {
            const int s = r + hi * 8;
            float x = (acc[r] + b1) * sc + sh;
            wg1[p][s][g] = x > 0.f ? x : 0.f;
        }
    }
    __syncthreads();

    // ---- wg2 = wg1 @ Ww2 + bw2 : tiny 16x16x16 per point, VALU ----
#pragma unroll
    for (int p = 0; p < 2; ++p) {
        const int s = tid >> 4, g = tid & 15;        // 256 threads = 16x16
        float x = bw2[g];
#pragma unroll
        for (int j = 0; j < GRP; ++j) x += wg1[p][s][j] * Ww2[j * GRP + g];
        wg2[p][s][g] = x;
    }
    __syncthreads();

    // ---- softmax over neighbors (axis S) per (point, group), then mask ----
    if (tid < 2 * GRP) {
        const int p = tid >> 4, g = tid & 15;
        float mx = -3.0e38f;
#pragma unroll
        for (int s = 0; s < SNB; ++s) mx = fmaxf(mx, wg2[p][s][g]);
        float ex[SNB], sum = 0.f;
#pragma unroll
        for (int s = 0; s < SNB; ++s) { ex[s] = __expf(wg2[p][s][g] - mx); sum += ex[s]; }
        const float inv = 1.f / sum;
#pragma unroll
        for (int s = 0; s < SNB; ++s) wg2[p][s][g] = ex[s] * inv * maskS[p][s];
    }
    __syncthreads();

    // ---- out[n, c] = sum_s vg[s][c] * wg2[s][c>>4] ----
#pragma unroll
    for (int p = 0; p < 2; ++p) {
        const int g = tid >> 4;                      // C/G == 16
        float acc = 0.f;
#pragma unroll
        for (int s = 0; s < SNB; ++s) acc += vg[p][s][tid] * wg2[p][s][g];
        out[(size_t)(n0 + p) * CCH + tid] = acc;
    }
}

// ---------------------------------------------------------------------------
extern "C" void kernel_launch(void* const* d_in, const int* in_sizes, int n_in,
                              void* d_out, int out_size, void* d_ws, size_t ws_size,
                              hipStream_t stream) {
    const float* feat  = (const float*)d_in[0];
    const float* coord = (const float*)d_in[1];
    const float* Wq = (const float*)d_in[2];  const float* bq = (const float*)d_in[3];
    const float* qs = (const float*)d_in[4];  const float* qh = (const float*)d_in[5];
    const float* Wk = (const float*)d_in[6];  const float* bk = (const float*)d_in[7];
    const float* ks = (const float*)d_in[8];  const float* kh = (const float*)d_in[9];
    const float* Wv = (const float*)d_in[10]; const float* bv = (const float*)d_in[11];
    const float* Wp1 = (const float*)d_in[12]; const float* bp1 = (const float*)d_in[13];
    const float* ps  = (const float*)d_in[14]; const float* ph  = (const float*)d_in[15];
    const float* Wp2 = (const float*)d_in[16]; const float* bp2 = (const float*)d_in[17];
    const float* Ww1 = (const float*)d_in[18]; const float* bw1 = (const float*)d_in[19];
    const float* wsc = (const float*)d_in[20]; const float* wsh = (const float*)d_in[21];
    const float* Ww2 = (const float*)d_in[22]; const float* bw2 = (const float*)d_in[23];
    const int*   ridx = (const int*)d_in[24];

    float* out = (float*)d_out;

    // Workspace layout:
    //   [0, 520KB)     : packed bf16 weights (Wq,Wk,Wv,Wp2: 128KB each; Ww1: 8KB)
    //   [1MB,  17MB)   : q fp32
    //   [17MB, 33MB)   : v fp32
    //   [33MB, 41MB)   : k bf16
    __bf16* pWq  = (__bf16*)d_ws;
    __bf16* pWk  = pWq  + 65536;
    __bf16* pWv  = pWk  + 65536;
    __bf16* pWp2 = pWv  + 65536;
    __bf16* pWw1 = pWp2 + 65536;
    float*  qbuf = (float*)((char*)d_ws + (1u << 20));
    float*  vbuf = qbuf + (size_t)NPTS * CCH;
    __bf16* kbuf = (__bf16*)(vbuf + (size_t)NPTS * CCH);

    pack_weight_kernel<<<256, 256, 0, stream>>>(Wq,  pWq,  CCH);
    pack_weight_kernel<<<256, 256, 0, stream>>>(Wk,  pWk,  CCH);
    pack_weight_kernel<<<256, 256, 0, stream>>>(Wv,  pWv,  CCH);
    pack_weight_kernel<<<256, 256, 0, stream>>>(Wp2, pWp2, CCH);
    pack_weight_kernel<<<16,  256, 0, stream>>>(Ww1, pWw1, GRP);

    qkv_gemm_kernel<<<NPTS / 32, 256, 0, stream>>>(
        feat, pWq, pWk, pWv, bq, qs, qh, bk, ks, kh, bv, qbuf, kbuf, vbuf);

    attn_fused_kernel<<<NPTS / 2, 256, 0, stream>>>(
        qbuf, kbuf, vbuf, coord, Wp1, bp1, ps, ph, pWp2, bp2,
        pWw1, bw1, wsc, wsh, Ww2, bw2, ridx, out);
}
